// ISTFTConv1D_27771258536451
// MI455X (gfx1250) — matrix operand order repaired and checked by reading, not compiled, not run on hardware
//
#include <hip/hip_runtime.h>

typedef __attribute__((ext_vector_type(16))) _Float16 v16h;
typedef __attribute__((ext_vector_type(8)))  _Float16 v8h;
typedef __attribute__((ext_vector_type(8)))  float    v8f;

#define NB     4096
#define NC     1026
#define CPAD   1056         // 33 * 32
#define NT     9
#define WIN    1024
#define HOP    256
#define NOUT   3072
#define NKT    33           // CPAD / 32

// ---------------------------------------------------------------------------
// Pack x [B,1026,9] f32  ->  Xp [9][B][1056] f16 (zero-padded c in [1026,1056))
// LDS transpose: contiguous global reads (c*9+t runs), coalesced f16 writes.
// grid (128 b-tiles, 33 c-tiles), 256 threads
// ---------------------------------------------------------------------------
__global__ __launch_bounds__(256) void pack_x_kernel(
    const float* __restrict__ x, _Float16* __restrict__ Xp)
{
    __shared__ float sx[32][292];               // [bb][cc*9+t], 288 used + pad
    const int b0 = blockIdx.x * 32;
    const int c0 = blockIdx.y * 32;
    const int tid = threadIdx.x;
    const int lim = NC * NT - c0 * NT;          // valid "off" range for this c-tile
    for (int f = tid; f < 32 * 288; f += 256) {
        int bb = f / 288, off = f - bb * 288;
        float v = 0.0f;
        if (off < lim)
            v = x[(size_t)(b0 + bb) * (NC * NT) + c0 * NT + off];
        sx[bb][off] = v;
    }
    __syncthreads();
    for (int g = tid; g < NT * 32 * 32; g += 256) {
        int cc = g & 31;
        int rest = g >> 5;
        int bb = rest & 31;
        int t  = rest >> 5;
        Xp[((size_t)t * NB + (b0 + bb)) * CPAD + c0 + cc] =
            (_Float16)sx[bb][cc * 9 + t];
    }
}

// ---------------------------------------------------------------------------
// Pack weight [1026,1024] f32 -> Wp [1024][1056] f16 (transposed, zero-padded)
// grid (33 c-tiles, 16 n-tiles), 256 threads
// ---------------------------------------------------------------------------
__global__ __launch_bounds__(256) void pack_w_kernel(
    const float* __restrict__ w, _Float16* __restrict__ Wp)
{
    __shared__ float sw[32][65];
    const int c0 = blockIdx.x * 32;
    const int n0 = blockIdx.y * 64;
    const int tid = threadIdx.x;
    for (int f = tid; f < 32 * 64; f += 256) {
        int cc = f >> 6, nn = f & 63;
        float v = (c0 + cc < NC) ? w[(size_t)(c0 + cc) * WIN + n0 + nn] : 0.0f;
        sw[cc][nn] = v;
    }
    __syncthreads();
    for (int g = tid; g < 32 * 64; g += 256) {
        int cc = g & 31, nn = g >> 5;
        Wp[(size_t)(n0 + nn) * CPAD + c0 + cc] = (_Float16)sw[cc][nn];
    }
}

// ---------------------------------------------------------------------------
// Fused GEMM + overlap-add + normalize.
//   y[b, j*256+q] = (sum_{d, t=j-d in [0,9)} sum_c Xp[t][b][c]*Wp[d*256+q][c])
//                   / max(ola[j*256+q], 1e-8)
// Block tile 128(b) x 256(q) = one full j column block; 8 waves (2m x 4n);
// wave tile 64 x 64 = 4x4 WMMA fragments (16 wmma per 16 ds_load_b128).
// Valid d segments are contiguous -> incrementally advanced prefetch pointers
// (no scratch arrays, no division). Global->register pipeline hides latency.
// grid (32 m-blocks, 12 j-blocks), 256 threads
// ---------------------------------------------------------------------------
__global__ __launch_bounds__(256) void istft_wmma_kernel(
    const _Float16* __restrict__ Xp,   // [9][4096][1056]
    const _Float16* __restrict__ Wp,   // [1024][1056]
    const float*    __restrict__ ola,  // [3072]
    float*          __restrict__ y)    // [4096][3072]
{
    __shared__ _Float16 lA[128][40];   // [m][k] row stride 80B (16B aligned)
    __shared__ _Float16 lB[256][40];   // [n][k]

    const int j    = blockIdx.y;            // output 256-column block (slowest)
    const int b0   = blockIdx.x * 128;      // batch tile base
    const int tid  = threadIdx.x;
    const int lane = tid & 31;
    const int wid  = tid >> 5;
    const int wm   = wid & 1;               // 2 waves along m (64 rows each)
    const int wn   = wid >> 1;              // 4 waves along n (64 cols each)
    const int lm   = lane & 15;
    const int lhi  = lane >> 4;

    const int cprow = tid >> 1;             // A copy: row 0..127
    const int cpseg = (tid & 1) * 16;       // A copy: halves 0 or 16 (32B each)

    // Contiguous valid d range: d in [dlo, dhi], t = j - d in [0, NT).
    int dlo = j - (NT - 1); if (dlo < 0) dlo = 0;
    int dhi = (j < 3) ? j : 3;
    const int ntiles = (dhi - dlo + 1) * NKT;

    // Prefetch pointers for segment dlo; advancing a segment means
    // t -= 1 (pA -= NB*CPAD) and d += 1 (pB += HOP*CPAD).
    const _Float16* pA = Xp + ((size_t)(j - dlo) * NB + b0 + cprow) * CPAD + cpseg;
    const _Float16* pB = Wp + (size_t)(dlo * HOP + tid) * CPAD;
    int kNext = 0;                          // k offset (halves) of next fetch

    v8h ra0 = ((const v8h*)pA)[0];
    v8h ra1 = ((const v8h*)pA)[1];
    v8h rb0 = ((const v8h*)pB)[0];
    v8h rb1 = ((const v8h*)pB)[1];
    v8h rb2 = ((const v8h*)pB)[2];
    v8h rb3 = ((const v8h*)pB)[3];

    v8f acc[4][4] = {};

    for (int it = 0; it < ntiles; ++it) {
        __syncthreads();                    // previous tile's frag reads done
        *(v8h*)&lA[cprow][cpseg]     = ra0;
        *(v8h*)&lA[cprow][cpseg + 8] = ra1;
        *(v8h*)&lB[tid][0]  = rb0;
        *(v8h*)&lB[tid][8]  = rb1;
        *(v8h*)&lB[tid][16] = rb2;
        *(v8h*)&lB[tid][24] = rb3;
        __syncthreads();

        if (it + 1 < ntiles) {              // prefetch next tile during compute
            kNext += 32;
            if (kNext == CPAD) {
                kNext = 0;
                pA -= (size_t)NB * CPAD;
                pB += (size_t)HOP * CPAD;
            }
            const v8h* sA = (const v8h*)(pA + kNext);
            ra0 = sA[0];
            ra1 = sA[1];
            const v8h* sB = (const v8h*)(pB + kNext);
            rb0 = sB[0];
            rb1 = sB[1];
            rb2 = sB[2];
            rb3 = sB[3];
        }

        // Fragment loads per documented 16-bit WMMA VGPR layouts.
        v16h a[4], b[4];
#pragma unroll
        for (int fm = 0; fm < 4; ++fm) {
            const _Float16* pa = &lA[wm * 64 + fm * 16 + lm][lhi * 8];
            v8h alo = *(const v8h*)pa;          // k = lhi*8 .. +7
            v8h ahi = *(const v8h*)(pa + 16);   // k = 16+lhi*8 .. +7
            a[fm] = __builtin_shufflevector(alo, ahi,
                    0,1,2,3,4,5,6,7,8,9,10,11,12,13,14,15);
        }
#pragma unroll
        for (int fn = 0; fn < 4; ++fn) {
            const _Float16* pb = &lB[wn * 64 + fn * 16 + lm][lhi * 8];
            v8h blo = *(const v8h*)pb;          // k = lhi*8 .. +7
            v8h bhi = *(const v8h*)(pb + 16);   // k = 16+lhi*8 .. +7
            b[fn] = __builtin_shufflevector(blo, bhi,
                    0,1,2,3,4,5,6,7,8,9,10,11,12,13,14,15);
        }
#pragma unroll
        for (int fm = 0; fm < 4; ++fm)
#pragma unroll
            for (int fn = 0; fn < 4; ++fn)
                acc[fm][fn] = __builtin_amdgcn_wmma_f32_16x16x32_f16(
                    false, a[fm], false, b[fn],
                    (short)0, acc[fm][fn], false, false);
    }

    // Epilogue: normalize and store (y written exactly once per element).
#pragma unroll
    for (int fn = 0; fn < 4; ++fn) {
        const int q = wn * 64 + fn * 16 + lm;   // 0..255 within j-block
        const int p = j * HOP + q;              // 0..3071
        const float rn = 1.0f / fmaxf(ola[p], 1e-8f);
#pragma unroll
        for (int fm = 0; fm < 4; ++fm) {
            const int mg = b0 + wm * 64 + fm * 16 + lhi * 8;
#pragma unroll
            for (int r = 0; r < 8; ++r)
                y[(size_t)(mg + r) * NOUT + p] = acc[fm][fn][r] * rn;
        }
    }
}

// ---------------------------------------------------------------------------
extern "C" void kernel_launch(void* const* d_in, const int* in_sizes, int n_in,
                              void* d_out, int out_size, void* d_ws, size_t ws_size,
                              hipStream_t stream) {
    const float* x   = (const float*)d_in[0];   // [4096, 1026, 9]
    const float* w   = (const float*)d_in[1];   // [1026, 1024]
    const float* ola = (const float*)d_in[2];   // [3072]
    float* y = (float*)d_out;                   // [4096, 3072]

    _Float16* Xp = (_Float16*)d_ws;                                   // 77,856,768 B
    _Float16* Wp = (_Float16*)((char*)d_ws + (size_t)NT * NB * CPAD * 2);

    pack_x_kernel<<<dim3(NB / 32, NKT), 256, 0, stream>>>(x, Xp);
    pack_w_kernel<<<dim3(NKT, WIN / 64), 256, 0, stream>>>(w, Wp);
    istft_wmma_kernel<<<dim3(NB / 128, NOUT / HOP), 256, 0, stream>>>(Xp, Wp, ola, y);
}
